// SelfAttention_transformer_v1_65395172049597
// MI455X (gfx1250) — matrix operand order repaired and checked by reading, not compiled, not run on hardware
//
#include <hip/hip_runtime.h>

// ---------------------------------------------------------------------------
// MI455X (gfx1250) fused self-attention pipeline, bf16 WMMA GEMMs + f32 accum
// ---------------------------------------------------------------------------

typedef __bf16 bf16;
typedef __attribute__((ext_vector_type(16))) __bf16 v16bf;
typedef __attribute__((ext_vector_type(8)))  __bf16 v8bf;
typedef __attribute__((ext_vector_type(8)))  float   v8f;
typedef __attribute__((ext_vector_type(4)))  float   v4f;

#define B_ 8
#define S_ 2048
#define D_ 1024

static __device__ __forceinline__ bf16 f2bf(float x) {
  union { float f; unsigned u; } v; v.f = x;
  unsigned r = v.u + 0x7FFFu + ((v.u >> 16) & 1u);  // round-to-nearest-even
  union { unsigned short u; bf16 b; } o; o.u = (unsigned short)(r >> 16);
  return o.b;
}

// --------------------------- f32 -> bf16 convert ---------------------------
__global__ __launch_bounds__(256) void cvt_bf16_kernel(const float* __restrict__ in,
                                                       bf16* __restrict__ out) {
  size_t i = ((size_t)blockIdx.x * blockDim.x + threadIdx.x) * 8;
  v8bf o;
#pragma unroll
  for (int r = 0; r < 8; ++r) o[r] = f2bf(in[i + r]);
  *(v8bf*)(out + i) = o;
}

// ------------------------------- zero output -------------------------------
__global__ __launch_bounds__(256) void zero_kernel(float* __restrict__ p) {
  p[(size_t)blockIdx.x * blockDim.x + threadIdx.x] = 0.0f;
}

// ---------------------------------------------------------------------------
// GEMM: C[M,N] = A[M,K] * Bt[N,K]^T (+bias).  A,Bt bf16 row-major.
// Block tile 128x128, staged K-depth 64 (two WMMA k-steps per barrier pair).
// 8 waves (wave32): 4 along M x 2 along N; each wave owns a 32x64 patch =
// 2x4 v_wmma_f32_16x16x32_bf16 accumulators, issued 16x per LDS stage.
// Fragment layouts per CDNA5 ISA 7.12.2:
//   A (16x32 bf16): lane L -> m=L%16, K chunks [8h,8h+8) and [16+8h,16+8h+8)
//   B (32x16 bf16): lane L -> n=L%16, K chunks [8h,8h+8) and [16+8h,16+8h+8)   (h = L/16)
// requires M%128==0, N%128==0, K%64==0 (true for all call sites here).
// ---------------------------------------------------------------------------
#define TM 128
#define TN 128
#define TK 64
#define LDS_P 72  // padded row stride (bf16) = 144 B: staggers banks, keeps 16B align

enum { OUT_F32 = 0, OUT_BF16 = 1, OUT_BF16_TRANS = 2 };

__global__ __launch_bounds__(256)
void gemm_bt_wmma_kernel(const bf16* __restrict__ A, int lda,
                         const bf16* __restrict__ Bt, int ldb,
                         void* __restrict__ C, int ldc,
                         int K, const float* __restrict__ bias, int flags) {
  __shared__ bf16 sA[TM * LDS_P];  // 18 KB
  __shared__ bf16 sB[TN * LDS_P];  // 18 KB

  const int tid   = threadIdx.x;
  const int wave  = tid >> 5;
  const int lane  = tid & 31;
  const int wm    = wave & 3;   // wave M coord (0..3)
  const int wn    = wave >> 2;  // wave N coord (0..1)
  const int l16   = lane & 15;
  const int lhalf = lane >> 4;
  const int mtile = blockIdx.y * TM;
  const int ntile = blockIdx.x * TN;

  v8f acc[2][4] = {};  // zero-initialized f32 accumulators

  for (int k0 = 0; k0 < K; k0 += TK) {
    // ---- stage A tile (128x64) and B tile (128x64, [N][K]) into LDS ----
    // 1024 16-byte chunks per tile, 4 per thread per tile.
#pragma unroll
    for (int q = 0; q < 4; ++q) {
      const int c = tid + q * 256;
      const int r = c >> 3, o8 = (c & 7) << 3;
      v8bf a = *(const v8bf*)(A + (size_t)(mtile + r) * lda + k0 + o8);
      v8bf b = *(const v8bf*)(Bt + (size_t)(ntile + r) * ldb + k0 + o8);
      *(v8bf*)&sA[r * LDS_P + o8] = a;
      *(v8bf*)&sB[r * LDS_P + o8] = b;
    }
    __syncthreads();

#pragma unroll
    for (int kk = 0; kk < 2; ++kk) {       // two 16x16x32 k-steps per stage
      const int kofs = kk * 32;
#pragma unroll
      for (int im = 0; im < 2; ++im) {
        const int arow = wm * 32 + im * 16 + l16;
        union { v16bf v; v8bf h[2]; } af;
        af.h[0] = *(const v8bf*)&sA[arow * LDS_P + kofs + 8 * lhalf];
        af.h[1] = *(const v8bf*)&sA[arow * LDS_P + kofs + 16 + 8 * lhalf];
#pragma unroll
        for (int in = 0; in < 4; ++in) {
          const int brow = wn * 64 + in * 16 + l16;
          union { v16bf v; v8bf h[2]; } bfr;
          bfr.h[0] = *(const v8bf*)&sB[brow * LDS_P + kofs + 8 * lhalf];
          bfr.h[1] = *(const v8bf*)&sB[brow * LDS_P + kofs + 16 + 8 * lhalf];
          acc[im][in] = __builtin_amdgcn_wmma_f32_16x16x32_bf16(
              false, af.v, false, bfr.v, (short)0, acc[im][in], false, false);
        }
      }
    }
    __syncthreads();
  }

  // ------------------------------ epilogue ---------------------------------
  // C/D layout: lane L -> n = L%16, VGPR r -> m = r + 8*(L/16).
#pragma unroll
  for (int im = 0; im < 2; ++im) {
#pragma unroll
    for (int in = 0; in < 4; ++in) {
      const int n = ntile + wn * 64 + in * 16 + l16;
      const float bv = bias ? bias[n] : 0.0f;
      const int mbase = mtile + wm * 32 + im * 16 + 8 * lhalf;
      v8f c = acc[im][in];
      if (flags == OUT_F32) {
        float* Cp = (float*)C;
#pragma unroll
        for (int r = 0; r < 8; ++r) Cp[(size_t)(mbase + r) * ldc + n] = c[r] + bv;
      } else if (flags == OUT_BF16) {
        bf16* Cp = (bf16*)C;
#pragma unroll
        for (int r = 0; r < 8; ++r) Cp[(size_t)(mbase + r) * ldc + n] = f2bf(c[r] + bv);
      } else {  // OUT_BF16_TRANS: Ct[n*ldc + m] -- contiguous 16B per lane
        bf16* Cp = (bf16*)C;
        v8bf pk;
#pragma unroll
        for (int r = 0; r < 8; ++r) pk[r] = f2bf(c[r] + bv);
        *(v8bf*)&Cp[(size_t)n * ldc + mbase] = pk;
      }
    }
  }
}

// -------------------- column softmax stats (axis=1 softmax) ----------------
// For each key column j: m[j] = max_i scores[i,j], s[j] = sum_i exp(x - m[j]).
__global__ __launch_bounds__(256)
void colstats_kernel(const float* __restrict__ scores, float* __restrict__ mcol,
                     float* __restrict__ scol) {
  const int j = blockIdx.x * blockDim.x + threadIdx.x;  // coalesced over j
  float m = -3.4e38f, s = 0.0f;
  for (int i = 0; i < S_; ++i) {
    float x = scores[(size_t)i * S_ + j];
    float mn = fmaxf(m, x);
    s = s * __expf(m - mn) + __expf(x - mn);
    m = mn;
  }
  mcol[j] = m;
  scol[j] = s;
}

// ------------- attn = exp(scores - m[j]) / s[j] * (1/sqrt(D)) --------------
__global__ __launch_bounds__(256)
void attn_kernel(const float* __restrict__ scores, const float* __restrict__ mcol,
                 const float* __restrict__ scol, bf16* __restrict__ attn, float scale) {
  size_t base = ((size_t)blockIdx.x * blockDim.x + threadIdx.x) * 8;
  const int j = (int)(base & (S_ - 1));  // 8 | S_, so j..j+7 share one row
  v8bf o;
#pragma unroll
  for (int r = 0; r < 8; ++r) {
    float x = scores[base + r];
    o[r] = f2bf(__expf(x - mcol[j + r]) / scol[j + r] * scale);
  }
  *(v8bf*)(attn + base) = o;
}

// ---------------- LayerNorm per row + mean-pool over sequence --------------
__global__ __launch_bounds__(256)
void lnpool_kernel(const float* __restrict__ x, const float* __restrict__ gamma,
                   const float* __restrict__ beta, float* __restrict__ out,
                   float inv_s) {
  __shared__ float red[256];
  const int t = threadIdx.x;
  const float* row = x + (size_t)blockIdx.x * D_;
  v4f v = *(const v4f*)(row + t * 4);

  red[t] = v[0] + v[1] + v[2] + v[3];
  __syncthreads();
  for (int o = 128; o > 0; o >>= 1) { if (t < o) red[t] += red[t + o]; __syncthreads(); }
  const float mu = red[0] * (1.0f / D_);
  __syncthreads();

  float d0 = v[0] - mu, d1 = v[1] - mu, d2 = v[2] - mu, d3 = v[3] - mu;
  red[t] = d0 * d0 + d1 * d1 + d2 * d2 + d3 * d3;
  __syncthreads();
  for (int o = 128; o > 0; o >>= 1) { if (t < o) red[t] += red[t + o]; __syncthreads(); }
  const float rs = rsqrtf(red[0] * (1.0f / D_) + 1e-5f);

#pragma unroll
  for (int r = 0; r < 4; ++r) {
    const int d = t * 4 + r;
    float val = (v[r] - mu) * rs * gamma[d] + beta[d];
    atomicAdd(&out[d], val * inv_s);
  }
}

// ---------------------------------------------------------------------------
extern "C" void kernel_launch(void* const* d_in, const int* in_sizes, int n_in,
                              void* d_out, int out_size, void* d_ws, size_t ws_size,
                              hipStream_t stream) {
  (void)in_sizes; (void)n_in; (void)out_size; (void)ws_size;
  const float* inp   = (const float*)d_in[0];
  // d_in[1] = lens: unused by the reference computation.
  const float* Wq    = (const float*)d_in[2];
  const float* bq    = (const float*)d_in[3];
  const float* Wk    = (const float*)d_in[4];
  const float* bk    = (const float*)d_in[5];
  const float* Wv    = (const float*)d_in[6];
  const float* bv    = (const float*)d_in[7];
  const float* gamma = (const float*)d_in[8];
  const float* beta  = (const float*)d_in[9];
  float* out = (float*)d_out;

  // ---- workspace layout (~86 MB): per-batch big intermediates stay in L2 --
  char* ws = (char*)d_ws;
  size_t off = 0;
  bf16* inpb = (bf16*)(ws + off); off += (size_t)B_ * S_ * D_ * 2;   // 32 MB
  bf16* wqb  = (bf16*)(ws + off); off += (size_t)D_ * D_ * 2;        //  2 MB
  bf16* wkb  = (bf16*)(ws + off); off += (size_t)D_ * D_ * 2;
  bf16* wvb  = (bf16*)(ws + off); off += (size_t)D_ * D_ * 2;
  bf16* Qb   = (bf16*)(ws + off); off += (size_t)S_ * D_ * 2;        //  4 MB
  bf16* Kb   = (bf16*)(ws + off); off += (size_t)S_ * D_ * 2;
  bf16* Vt   = (bf16*)(ws + off); off += (size_t)D_ * S_ * 2;        // V^T [D,S]
  float* scores = (float*)(ws + off); off += (size_t)S_ * S_ * 4;    // 16 MB
  bf16*  attnb  = (bf16*)(ws + off);  off += (size_t)S_ * S_ * 2;    //  8 MB
  float* mcol   = (float*)(ws + off); off += (size_t)S_ * 4;
  float* scol   = (float*)(ws + off); off += (size_t)S_ * 4;
  float* selfat = (float*)(ws + off); off += (size_t)S_ * D_ * 4;    //  8 MB

  const float scale = 0.03125f;        // 1/sqrt(1024)
  const float inv_s = 1.0f / (float)S_;

  // ---- one-time converts ----
  cvt_bf16_kernel<<<(size_t)B_ * S_ * D_ / (256 * 8), 256, 0, stream>>>(inp, inpb);
  cvt_bf16_kernel<<<(size_t)D_ * D_ / (256 * 8), 256, 0, stream>>>(Wq, wqb);
  cvt_bf16_kernel<<<(size_t)D_ * D_ / (256 * 8), 256, 0, stream>>>(Wk, wkb);
  cvt_bf16_kernel<<<(size_t)D_ * D_ / (256 * 8), 256, 0, stream>>>(Wv, wvb);
  zero_kernel<<<B_ * D_ / 256, 256, 0, stream>>>(out);

  const dim3 gQKV(D_ / TN, S_ / TM);   // (8,16)
  const dim3 gSS (S_ / TN, S_ / TM);   // (16,16)

  for (int b = 0; b < B_; ++b) {
    const bf16* xb = inpb + (size_t)b * S_ * D_;
    // Q = x*Wq^T+bq ; K = x*Wk^T+bk ; V^T (transposed epilogue) = x*Wv^T+bv
    gemm_bt_wmma_kernel<<<gQKV, 256, 0, stream>>>(xb, D_, wqb, D_, Qb, D_, D_, bq, OUT_BF16);
    gemm_bt_wmma_kernel<<<gQKV, 256, 0, stream>>>(xb, D_, wkb, D_, Kb, D_, D_, bk, OUT_BF16);
    gemm_bt_wmma_kernel<<<gQKV, 256, 0, stream>>>(xb, D_, wvb, D_, Vt, S_, D_, bv, OUT_BF16_TRANS);
    // scores[i,j] = Q[i,:].K[j,:]
    gemm_bt_wmma_kernel<<<gSS, 256, 0, stream>>>(Qb, D_, Kb, D_, scores, S_, D_, nullptr, OUT_F32);
    // softmax over the QUERY axis (per-column j), then /sqrt(D)
    colstats_kernel<<<S_ / 256, 256, 0, stream>>>(scores, mcol, scol);
    attn_kernel<<<(size_t)S_ * S_ / (256 * 8), 256, 0, stream>>>(scores, mcol, scol, attnb, scale);
    // selfatt = attn * V   (B^T = V^T [D,S])
    gemm_bt_wmma_kernel<<<gQKV, 256, 0, stream>>>(attnb, S_, Vt, S_, selfat, D_, S_, nullptr, OUT_F32);
    // LayerNorm rows + mean-pool into out[b,:]
    lnpool_kernel<<<S_, 256, 0, stream>>>(selfat, gamma, beta, out + (size_t)b * D_, inv_s);
  }
}
